// SequenceEncoder_40802189312829
// MI455X (gfx1250) — compile-verified, matrix operand order checked by reading, not run on hardware
//
#include <hip/hip_runtime.h>
#include <hip/hip_bf16.h>

typedef __bf16 bf16_t;
typedef bf16_t v16bf __attribute__((ext_vector_type(16)));
typedef float  v8f   __attribute__((ext_vector_type(8)));

#define Bn 32
#define Tn 1024
#define Hn 256
#define HPAD 264          // padded LDS row stride (bf16 elems): 528B -> lane stride 4 banks
#define DMODEL 512        // 2H
#define OUTC 513          // 2H+1
#define LOG2E 1.4426950408889634f

// Hardware transcendentals: gfx1250 has v_tanh_f32 (TRANS op, co-executes with WMMA).
__device__ __forceinline__ float fast_tanh(float x) {
#if __has_builtin(__builtin_amdgcn_tanhf)
    return __builtin_amdgcn_tanhf(x);
#elif __has_builtin(__builtin_amdgcn_tanh_f32)
    return __builtin_amdgcn_tanh_f32(x);
#else
    // tanh(x) = (1 - e^-2x) / (1 + e^-2x) via v_exp_f32 (exp2) + v_rcp_f32
    float e = __builtin_amdgcn_exp2f(x * (-2.0f * LOG2E));
    return (1.0f - e) * __builtin_amdgcn_rcpf(1.0f + e);
#endif
}
__device__ __forceinline__ float fast_sigmoid(float x) {
    return fmaf(fast_tanh(0.5f * x), 0.5f, 0.5f);
}

// ---------------------------------------------------------------------------
// 1) zscore (ddof=1) + lagged diff; also writes out[..., 512] = x
// ---------------------------------------------------------------------------
__global__ __launch_bounds__(1024) void preprocess_kernel(const float* __restrict__ batch,
                                                          float* __restrict__ x,
                                                          float* __restrict__ out)
{
    __shared__ float red[Tn];
    __shared__ float xz[Tn];
    const int b = blockIdx.x, tid = threadIdx.x;
    float v = batch[b * Tn + tid];
    red[tid] = v; __syncthreads();
    for (int s = Tn / 2; s > 0; s >>= 1) { if (tid < s) red[tid] += red[tid + s]; __syncthreads(); }
    float mean = red[0] / (float)Tn;
    __syncthreads();
    float d = v - mean;
    red[tid] = d * d; __syncthreads();
    for (int s = Tn / 2; s > 0; s >>= 1) { if (tid < s) red[tid] += red[tid + s]; __syncthreads(); }
    float sd = sqrtf(red[0] / (float)(Tn - 1));
    float z = d / sd;
    xz[tid] = z; __syncthreads();
    float xv = (tid == 0) ? 0.0f : (z - xz[tid - 1]);
    x[b * Tn + tid] = xv;
    out[(size_t)(b * Tn + tid) * OUTC + DMODEL] = xv;
}

// ---------------------------------------------------------------------------
// 2) Pack W_hh^T (both dirs) into WMMA B-operand layout, bf16.
//    16-bit operand layout: lane -> {half = lane>>4, n = lane&15};
//    element e -> vgpr=e/2,pos=e%2; k = 16*(vgpr>>2) + 8*half + 2*(vgpr&3) + pos
// ---------------------------------------------------------------------------
__global__ void pack_whh_kernel(const float* __restrict__ Wf, const float* __restrict__ Wb,
                                bf16_t* __restrict__ wpf, bf16_t* __restrict__ wpb)
{
    int id = blockIdx.x * blockDim.x + threadIdx.x;   // 2 * 512 tiles * 512 elems
    int dir = id >> 18;                               // 262144 elems per direction
    int rem = id & 262143;
    int tile = rem >> 9;
    int within = rem & 511;
    int lane = within >> 4;
    int e = within & 15;
    int vg = e >> 1, pos = e & 1;
    int half = lane >> 4, nl = lane & 15;
    int k = 16 * (vg >> 2) + 8 * half + 2 * (vg & 3) + pos;
    int ks = tile >> 6, nt = tile & 63;
    int K = ks * 32 + k;          // hidden-state dim (0..255)
    int N = nt * 16 + nl;         // gate column   (0..1023)
    const float* W = dir ? Wb : Wf;       // W_hh row-major [4H][H]; B[k][n] = W[n][k]
    bf16_t* wp = dir ? wpb : wpf;
    wp[rem] = (bf16_t)W[N * Hn + K];
}

// ---------------------------------------------------------------------------
// 3) BiLSTM recurrence. grid=2 (direction), 512 threads = 16 waves.
//    Wave w owns hidden units [16w,16w+16): 4 gate N-tiles x 2 M-tiles.
//    64 v_wmma_f32_16x16x32_bf16 per wave per timestep.
//    Double-buffered h state in LDS -> single barrier per timestep; x column
//    prefetched one step ahead.
// ---------------------------------------------------------------------------
__global__ __launch_bounds__(512) void bilstm_kernel(
    const float* __restrict__ x,
    const float* __restrict__ Wih_f, const float* __restrict__ bias_f,
    const float* __restrict__ Wih_b, const float* __restrict__ bias_b,
    const bf16_t* __restrict__ wpf, const bf16_t* __restrict__ wpb,
    float* __restrict__ Hmat)
{
    __shared__ alignas(16) bf16_t hbuf[2][Bn * HPAD];   // double-buffered h, bf16
    __shared__ float xs[2][Bn];                         // double-buffered x column

    const int dir  = blockIdx.x;
    const int tid  = threadIdx.x;
    const int w    = tid >> 5;        // wave id 0..15
    const int lane = tid & 31;
    const int half = lane >> 4;
    const int nl   = lane & 15;
    const int unit = w * 16 + nl;     // hidden unit this lane owns in C/D layout

    const float*  Wih = dir ? Wih_b : Wih_f;
    const float*  bv  = dir ? bias_b : bias_f;
    const bf16_t* wp  = dir ? wpb : wpf;
    const int dirOff  = dir ? Hn : 0;

    float wih[4], bia[4];
#pragma unroll
    for (int g = 0; g < 4; ++g) { int col = g * Hn + unit; wih[g] = Wih[col]; bia[g] = bv[col]; }

    float creg[16];
#pragma unroll
    for (int i = 0; i < 16; ++i) creg[i] = 0.0f;

    for (int i = tid; i < Bn * HPAD; i += 512) hbuf[0][i] = (bf16_t)0.0f;
    if (tid < Bn) xs[0][tid] = x[tid * Tn + (dir ? (Tn - 1) : 0)];
    __syncthreads();

    int p = 0;
    for (int t = 0; t < Tn; ++t) {
        const int time = dir ? (Tn - 1 - t) : t;

        // prefetch next step's x column into the other buffer (off critical path)
        if (tid < Bn && (t + 1) < Tn) {
            const int tnext = dir ? (Tn - 2 - t) : (t + 1);
            xs[1 - p][tid] = x[tid * Tn + tnext];
        }

        // C init = input projection x_t * W_ih + b, directly in C/D layout:
        // row = m*16 + 8*half + r, col = unit
        v8f acc[4][2];
#pragma unroll
        for (int g = 0; g < 4; ++g)
#pragma unroll
            for (int m = 0; m < 2; ++m)
#pragma unroll
                for (int r = 0; r < 8; ++r) {
                    int row = m * 16 + half * 8 + r;
                    acc[g][m][r] = fmaf(xs[p][row], wih[g], bia[g]);
                }

#pragma unroll 2
        for (int ks = 0; ks < 8; ++ks) {
            // A operand: lane -> row = m*16 + (lane&15); K runs
            // [ks*32+8*half, +8) and [ks*32+16+8*half, +8): two b128 LDS loads.
            v16bf A[2];
#pragma unroll
            for (int m = 0; m < 2; ++m) {
                const bf16_t* ptr = &hbuf[p][(m * 16 + nl) * HPAD + ks * 32 + 8 * half];
                ((uint4*)&A[m])[0] = *(const uint4*)ptr;
                ((uint4*)&A[m])[1] = *(const uint4*)(ptr + 16);
            }
#pragma unroll
            for (int g = 0; g < 4; ++g) {
                const v16bf Bt = *(const v16bf*)(wp + ((size_t)((ks * 64 + (g * 16 + w)) * 32 + lane)) * 16);
#pragma unroll
                for (int m = 0; m < 2; ++m)
                    acc[g][m] = __builtin_amdgcn_wmma_f32_16x16x32_bf16(
                        false, A[m], false, Bt, (short)0, acc[g][m], false, false);
            }
        }

        // Gate nonlinearity (torch order i,f,g,o) via hardware TRANS ops;
        // c stays f32 in registers; h -> other LDS buffer (no WAR hazard) + Hmat.
#pragma unroll
        for (int m = 0; m < 2; ++m)
#pragma unroll
            for (int r = 0; r < 8; ++r) {
                int row = m * 16 + half * 8 + r;
                float iv = fast_sigmoid(acc[0][m][r]);
                float fv = fast_sigmoid(acc[1][m][r]);
                float gv = fast_tanh(acc[2][m][r]);
                float ov = fast_sigmoid(acc[3][m][r]);
                float c  = fmaf(fv, creg[m * 8 + r], iv * gv);
                creg[m * 8 + r] = c;
                float hv = ov * fast_tanh(c);
                hbuf[1 - p][row * HPAD + unit] = (bf16_t)hv;
                Hmat[((size_t)row * Tn + time) * DMODEL + dirOff + unit] = hv;
            }

        __syncthreads();   // single barrier: publishes h(t) and xs(t+1) for next step
        p ^= 1;
    }
}

// ---------------------------------------------------------------------------
// 4) Attention: softmax over j of s_key[b,j] (query-invariant!) -> Hsum[b,512]
// ---------------------------------------------------------------------------
__global__ __launch_bounds__(1024) void attn_kernel(const float* __restrict__ Hmat,
                                                    const float* __restrict__ w_attn,
                                                    const float* __restrict__ b_attn,
                                                    float* __restrict__ Hsum)
{
    __shared__ float wk[DMODEL];
    __shared__ float skey[Tn];
    __shared__ float red[Tn];
    const int b = blockIdx.x, tid = threadIdx.x;
    if (tid < DMODEL) wk[tid] = w_attn[tid];
    __syncthreads();

    const int wv = tid >> 5, lane = tid & 31;
    const float battn = b_attn[0];
    for (int jj = 0; jj < 32; ++jj) {
        int j = wv * 32 + jj;
        const float* hp = Hmat + ((size_t)b * Tn + j) * DMODEL;
        float pacc = 0.f;
#pragma unroll
        for (int d = lane; d < DMODEL; d += 32) pacc = fmaf(hp[d], wk[d], pacc);
#pragma unroll
        for (int off = 16; off > 0; off >>= 1) pacc += __shfl_xor(pacc, off);
        if (lane == 0) skey[j] = pacc + battn;
    }
    __syncthreads();

    red[tid] = skey[tid]; __syncthreads();
    for (int s = Tn / 2; s > 0; s >>= 1) { if (tid < s) red[tid] = fmaxf(red[tid], red[tid + s]); __syncthreads(); }
    float mx = red[0]; __syncthreads();
    float e = __builtin_amdgcn_exp2f((skey[tid] - mx) * LOG2E);   // v_exp_f32
    red[tid] = e; __syncthreads();
    for (int s = Tn / 2; s > 0; s >>= 1) { if (tid < s) red[tid] += red[tid + s]; __syncthreads(); }
    float denom = red[0]; __syncthreads();
    skey[tid] = e * __builtin_amdgcn_rcpf(denom);
    __syncthreads();

    if (tid < DMODEL) {
        float a = 0.f;
        for (int j = 0; j < Tn; ++j) a = fmaf(skey[j], Hmat[((size_t)b * Tn + j) * DMODEL + tid], a);
        Hsum[b * DMODEL + tid] = a;
    }
}

// ---------------------------------------------------------------------------
// 5) Broadcast Hsum across T into out[..., 0:512]
// ---------------------------------------------------------------------------
__global__ void bcast_kernel(const float* __restrict__ Hsum, float* __restrict__ out)
{
    const int idx = blockIdx.x;        // b*Tn + t
    const int b = idx >> 10;
    const float* hs = Hsum + b * DMODEL;
    float* op = out + (size_t)idx * OUTC;
    for (int d = threadIdx.x; d < DMODEL; d += blockDim.x) op[d] = hs[d];
}

// ---------------------------------------------------------------------------
extern "C" void kernel_launch(void* const* d_in, const int* in_sizes, int n_in,
                              void* d_out, int out_size, void* d_ws, size_t ws_size,
                              hipStream_t stream)
{
    const float* batch  = (const float*)d_in[0];
    const float* Wih_f  = (const float*)d_in[1];
    const float* Whh_f  = (const float*)d_in[2];
    const float* b_f    = (const float*)d_in[3];
    const float* Wih_b  = (const float*)d_in[4];
    const float* Whh_b  = (const float*)d_in[5];
    const float* b_b    = (const float*)d_in[6];
    const float* w_attn = (const float*)d_in[7];
    const float* b_attn = (const float*)d_in[8];
    float* out = (float*)d_out;

    char* ws = (char*)d_ws;
    float*  x    = (float*)(ws + 0);          // 32*1024*4       = 131072 B
    float*  Hsum = (float*)(ws + 131072);     // 32*512*4        = 65536 B
    bf16_t* wpf  = (bf16_t*)(ws + 196608);    // 512 tiles * 1KB = 524288 B
    bf16_t* wpb  = (bf16_t*)(ws + 720896);    // 524288 B
    float*  Hmat = (float*)(ws + 1245184);    // 32*1024*512*4   = 67108864 B

    preprocess_kernel<<<Bn, Tn, 0, stream>>>(batch, x, out);
    pack_whh_kernel<<<2048, 256, 0, stream>>>(Whh_f, Whh_b, wpf, wpb);
    bilstm_kernel<<<2, 512, 0, stream>>>(x, Wih_f, b_f, Wih_b, b_b, wpf, wpb, Hmat);
    attn_kernel<<<Bn, Tn, 0, stream>>>(Hmat, w_attn, b_attn, Hsum);
    bcast_kernel<<<Bn * Tn, 256, 0, stream>>>(Hsum, out);
}
